// HRWKV_7_90194313216252
// MI455X (gfx1250) — compile-verified, matrix operand order checked by reading, not run on hardware
//
#include <hip/hip_runtime.h>
#include <hip/hip_bf16.h>
#include <math.h>

#define TT   256
#define HH   32
#define NN   128
#define HNH  4096
#define KVHN 8
#define KVD  1024
#define KVREP 4
#define MT   8        // m-tiles per wave in GEMM (2 waves cover M=256)

typedef __attribute__((ext_vector_type(16))) __bf16 v16bf;
typedef __attribute__((ext_vector_type(8)))  __bf16 v8bf;
typedef __attribute__((ext_vector_type(8)))  float  v8f;

__device__ __forceinline__ unsigned short f2bf(float f) {
  unsigned int u = __float_as_uint(f);
  u += 0x7FFFu + ((u >> 16) & 1u);     // round-to-nearest-even
  return (unsigned short)(u >> 16);
}

__device__ __forceinline__ float block_reduce(float v, float* red, int n) {
  int i = threadIdx.x;
  red[i] = v; __syncthreads();
  for (int s = n >> 1; s > 0; s >>= 1) {
    if (i < s) red[i] += red[i + s];
    __syncthreads();
  }
  float r = red[0]; __syncthreads();
  return r;
}

// ---------------- RMSNorm over rows of HN, optional bf16 copy + last-row tap ----
__global__ void rmsnorm_rows(const float* __restrict__ in, const float* __restrict__ w,
                             float* __restrict__ out, unsigned short* __restrict__ outb,
                             float* __restrict__ xlast) {
  __shared__ float red[256];
  int t = blockIdx.x;
  const float* row = in + (size_t)t * HNH;
  float s = 0.f;
  for (int i = threadIdx.x; i < HNH; i += 256) { float v = row[i]; s += v * v; }
  float tot = block_reduce(s, red, 256);
  float rinv = rsqrtf(tot / (float)HNH + 1e-6f);
  for (int i = threadIdx.x; i < HNH; i += 256) {
    float v = row[i] * rinv * w[i];
    out[(size_t)t * HNH + i] = v;
    if (outb) outb[(size_t)t * HNH + i] = f2bf(v);
    if (xlast && t == TT - 1) xlast[i] = v;
  }
}

// ---------------- WMMA bf16 GEMM with 8 accumulator tiles per wave --------------
// Y[t][n] = sum_k Xb[t][k]*bf16(W[n][k]) (+bias[n]) (+resid[t][n])
// M = 256. Wave = one 16-col slab x half of M (8 m-tiles). 2 waves per slab.
// Per k-chunk: issue all 16 A loads in one clause, overlap with B fp32->bf16
// conversion, then 8 WMMAs drain with staggered loadcnt waits.
// grid = Nout/64 blocks of 256 threads (8 waves).
__global__ void gemm_bf16_wmma(const unsigned short* __restrict__ Xb,
                               const float* __restrict__ W,
                               const float* __restrict__ bias,
                               const float* __restrict__ resid,
                               float* __restrict__ Y, int Nout) {
  int wv   = threadIdx.x >> 5;
  int lane = threadIdx.x & 31;
  int slab = blockIdx.x * 4 + (wv >> 1);   // n-slab (16 cols)
  int mh   = wv & 1;                       // M half
  int n0 = slab * 16;
  int m0 = mh * 128;
  int g  = lane >> 4;                      // lane half (K interleave group)
  int mn = lane & 15;                      // A row / B col within tile

  const float*          wrow  = W  + (size_t)(n0 + mn) * HNH;
  const unsigned short* abase = Xb + (size_t)(m0 + mn) * HNH + (size_t)g * 8;

  v8f z = {};
  v8f acc[MT];
#pragma unroll
  for (int m = 0; m < MT; m++) acc[m] = z;

  for (int k0 = 0; k0 < HNH; k0 += 32) {
    __builtin_prefetch(wrow + k0 + 512, 0, 0);   // stream weights ahead

    // 1) launch ALL A-fragment loads for this k-chunk (one long clause)
    union Afrag { v16bf v; v8bf h[2]; };
    Afrag A[MT];
#pragma unroll
    for (int m = 0; m < MT; m++) {
      const unsigned short* arow = abase + (size_t)m * 16 * HNH + k0;
      A[m].h[0] = *(const v8bf*)(arow);
      A[m].h[1] = *(const v8bf*)(arow + 16);
    }

    // 2) B fp32 load + bf16 convert overlaps the A loads above
    v8f wlo = *(const v8f*)(wrow + k0 + g * 8);
    v8f whi = *(const v8f*)(wrow + k0 + 16 + g * 8);
    union { v16bf v; unsigned short u[16]; } Bm;
#pragma unroll
    for (int e = 0; e < 8; e++) { Bm.u[e] = f2bf(wlo[e]); Bm.u[8 + e] = f2bf(whi[e]); }

    // 3) drain: WMMAs retire as their A fragments arrive
#pragma unroll
    for (int m = 0; m < MT; m++)
      acc[m] = __builtin_amdgcn_wmma_f32_16x16x32_bf16(false, A[m].v, false, Bm.v,
                                                       (short)0, acc[m], false, false);
  }

  int col = n0 + mn;
  float bv = bias ? bias[col] : 0.f;
  const size_t stride = (size_t)Nout;
  float*       ybase = Y + (size_t)m0 * stride + col;
  if (resid) {
    const float* rbase = resid + (size_t)m0 * stride + col;
#pragma unroll
    for (int m = 0; m < MT; m++)
#pragma unroll
      for (int i = 0; i < 8; i++) {
        size_t off = (size_t)(m * 16 + g * 8 + i) * stride;
        ybase[off] = acc[m][i] + bv + rbase[off];
      }
  } else {
#pragma unroll
    for (int m = 0; m < MT; m++)
#pragma unroll
      for (int i = 0; i < 8; i++) {
        size_t off = (size_t)(m * 16 + g * 8 + i) * stride;
        ybase[off] = acc[m][i] + bv;
      }
  }
}

// ---------------- per-head RMSNorm + NTK-mixed RoPE (in place), 128 threads -----
__global__ void headnorm_rope(float* __restrict__ buf, const float* __restrict__ lnw,
                              const int* __restrict__ cachepos, int heads) {
  __shared__ float red[128];
  __shared__ float sv[128];
  int t = blockIdx.x / heads, h = blockIdx.x % heads;
  float* row = buf + ((size_t)t * heads + h) * NN;
  int i = threadIdx.x;
  float v = row[i];
  float tot = block_reduce(v * v, red, 128);
  float xn = v * rsqrtf(tot / (float)NN + 1e-6f) * lnw[i];
  sv[i] = xn; __syncthreads();
  float pos = (float)cachepos[0] + (float)t;
  float e = (float)(i & 63) / 64.0f;
  float inv_short = powf(10000.0f, -e);
  float inv_long  = 1.0f / (10000.0f * powf(10.0f, e));
  float mix = fminf(fmaxf(pos / 4096.0f, 0.f), 1.f);
  float fr = (1.f - mix) * (pos * inv_short) + mix * (pos * inv_long);
  float rot = (i < 64) ? -sv[i + 64] : sv[i - 64];
  row[i] = xn * __cosf(fr) + rot * __sinf(fr);
}

// ---------------- LoRA stage 1:  tmp[t][n] = act(sum_k x[t][k]*W1[k][n]) ----------
__global__ void lora1(const float* __restrict__ x, const float* __restrict__ W1,
                      float* __restrict__ tmp, int Nl, int act) {
  int t = blockIdx.x, n = threadIdx.x;
  const float* xr = x + (size_t)t * HNH;
  float s = 0.f;
  for (int k = 0; k < HNH; k++) s += xr[k] * W1[(size_t)k * Nl + n];
  if (act == 1) s = tanhf(s);
  else if (act == 2) s = 1.f / (1.f + __expf(-s));
  tmp[t * Nl + n] = s;
}

// ---------------- LoRA stage 2:  out[t][n] = post(sum_l tmp[t][l]*W2[l][n] + b0[n])
// mode 0: w-transform (-softplus(-x)-0.5), 1: sigmoid, 2: none
__global__ void lora2(const float* __restrict__ tmp, const float* __restrict__ W2,
                      const float* __restrict__ b0, float* __restrict__ out,
                      int Nl, int mode) {
  __shared__ float tl[128];
  int t = blockIdx.x;
  int n = blockIdx.y * 256 + threadIdx.x;
  if (threadIdx.x < Nl) tl[threadIdx.x] = tmp[t * Nl + threadIdx.x];
  __syncthreads();
  float s = 0.f;
  for (int l = 0; l < Nl; l++) s += tl[l] * W2[(size_t)l * HNH + n];
  if (b0) s += b0[n];
  if (mode == 0) {
    float z = -s;
    float sp = (z > 20.f) ? z : log1pf(__expf(z));
    s = -sp - 0.5f;
  } else if (mode == 1) {
    s = 1.f / (1.f + __expf(-s));
  }
  out[(size_t)t * HNH + n] = s;
}

// ---------------- prep: repeat KV heads, v_first mix, kk norm, k*(1-w+a), decay --
__global__ void prep_scan(const float* __restrict__ kraw, const float* __restrict__ vraw,
                          const float* __restrict__ vfirst, const float* __restrict__ gate,
                          const float* __restrict__ wb, const float* __restrict__ ab,
                          float* __restrict__ kscan, float* __restrict__ vout,
                          float* __restrict__ kk, float* __restrict__ decay) {
  __shared__ float red[128];
  int t = blockIdx.x / HH, h = blockIdx.x % HH;
  int kvh = h / KVREP;
  int i = threadIdx.x;
  size_t gi = ((size_t)t * HH + h) * NN + i;
  float kv = kraw[((size_t)t * KVHN + kvh) * NN + i];
  float vv = vraw[((size_t)t * KVHN + kvh) * NN + i];
  float nrm2 = block_reduce(kv * kv, red, 128);
  float nrm = fmaxf(sqrtf(nrm2), 1e-12f);
  float w = wb[gi], a = ab[gi], gt = gate[gi];
  kk[gi]    = kv / nrm;
  kscan[gi] = kv * (1.f - w + a);
  vout[gi]  = vv + (vfirst[gi] - vv) * gt;
  decay[gi] = __expf(-__expf(w));
}

// ---------------- sequential scan: one block per head, thread i owns row S[i][:] --
__global__ void scan_kernel(const float* __restrict__ r, const float* __restrict__ decay,
                            const float* __restrict__ kscan, const float* __restrict__ vout,
                            const float* __restrict__ kk, const float* __restrict__ ab,
                            const float* __restrict__ S0, float* __restrict__ outs,
                            float* __restrict__ Sfin) {
  int h = blockIdx.x, i = threadIdx.x;
  float S[NN];
#pragma unroll
  for (int j = 0; j < NN; j++) S[j] = S0[((size_t)h * NN + i) * NN + j];
  __shared__ float sr[NN], sd[NN], sk[NN], sv[NN], sa[NN], sb[NN];
  for (int t = 0; t < TT; t++) {
    size_t gi = ((size_t)t * HH + h) * NN + i;
    float kkv = kk[gi];
    sr[i] = r[gi]; sd[i] = decay[gi]; sk[i] = kscan[gi]; sv[i] = vout[gi];
    sa[i] = -kkv;  sb[i] = kkv * ab[gi];
    __syncthreads();
    float sai = 0.f;
#pragma unroll
    for (int j = 0; j < NN; j++) sai += S[j] * sa[j];
    float vi = sv[i];
    float oi = 0.f;
#pragma unroll
    for (int j = 0; j < NN; j++) {
      float s = S[j] * sd[j] + sai * sb[j] + vi * sk[j];
      S[j] = s;
      oi += s * sr[j];
    }
    outs[gi] = oi;
    __syncthreads();
  }
#pragma unroll
  for (int j = 0; j < NN; j++) Sfin[((size_t)h * NN + i) * NN + j] = S[j];
}

// ---------------- bonus + gate + bf16 pack for O-GEMM ---------------------------
__global__ void bonus_gate(const float* __restrict__ outs, const float* __restrict__ r,
                           const float* __restrict__ kscan, const float* __restrict__ vout,
                           const float* __restrict__ rk, const float* __restrict__ g,
                           unsigned short* __restrict__ yb) {
  __shared__ float red[128];
  int t = blockIdx.x / HH, h = blockIdx.x % HH;
  int i = threadIdx.x;
  size_t gi = ((size_t)t * HH + h) * NN + i;
  float dot = block_reduce(r[gi] * kscan[gi] * rk[h * NN + i], red, 128);
  float xx = outs[gi] * 0.08838834764831845f + dot * vout[gi];   // N^-0.5 = 1/sqrt(128)
  yb[gi] = f2bf(xx * g[gi]);
}

extern "C" void kernel_launch(void* const* d_in, const int* in_sizes, int n_in,
                              void* d_out, int out_size, void* d_ws, size_t ws_size,
                              hipStream_t stream) {
  const float* x_in   = (const float*)d_in[0];
  const float* vfirst = (const float*)d_in[1];
  const float* state0 = (const float*)d_in[2];
  const int*   cpos   = (const int*)  d_in[3];
  const float* w0 = (const float*)d_in[4],  *w1 = (const float*)d_in[5],  *w2 = (const float*)d_in[6];
  const float* a0 = (const float*)d_in[7],  *a1 = (const float*)d_in[8],  *a2 = (const float*)d_in[9];
  const float* v0 = (const float*)d_in[10], *v1 = (const float*)d_in[11], *v2 = (const float*)d_in[12];
  const float* g1 = (const float*)d_in[13], *g2 = (const float*)d_in[14];
  const float* r_k = (const float*)d_in[15];
  const float* R_ = (const float*)d_in[16], *K_ = (const float*)d_in[17];
  const float* V_ = (const float*)d_in[18], *O_ = (const float*)d_in[19];
  const float* Rb = (const float*)d_in[20], *Kb = (const float*)d_in[21], *Vb = (const float*)d_in[22];
  const float* ln_r = (const float*)d_in[23], *ln_k = (const float*)d_in[24];
  const float* ln1  = (const float*)d_in[25], *ln2  = (const float*)d_in[26];

  // d_out layout (floats): out | x_last | S_final | v_first | x_res
  float* o_out   = (float*)d_out;
  float* o_xlast = o_out   + (size_t)TT * HNH;
  float* o_sfin  = o_xlast + HNH;
  float* o_vfst  = o_sfin  + (size_t)HH * NN * NN;
  float* o_xres  = o_vfst  + (size_t)TT * HNH;

  // workspace carve-up
  char* p = (char*)d_ws;
  auto take = [&](size_t bytes) { char* q = p; p += (bytes + 255) & ~(size_t)255; return q; };
  float*          x     = (float*)         take((size_t)TT * HNH * 4);
  unsigned short* xb    = (unsigned short*)take((size_t)TT * HNH * 2);
  float*          rbuf  = (float*)         take((size_t)TT * HNH * 4);
  float*          kraw  = (float*)         take((size_t)TT * KVD * 4);
  float*          vraw  = (float*)         take((size_t)TT * KVD * 4);
  float*          wbuf  = (float*)         take((size_t)TT * HNH * 4);
  float*          abuf  = (float*)         take((size_t)TT * HNH * 4);
  float*          gbuf  = (float*)         take((size_t)TT * HNH * 4);
  float*          gatbf = (float*)         take((size_t)TT * HNH * 4);
  float*          kscan = (float*)         take((size_t)TT * HNH * 4);
  float*          vbuf  = (float*)         take((size_t)TT * HNH * 4);
  float*          kkbuf = (float*)         take((size_t)TT * HNH * 4);
  float*          dcay  = (float*)         take((size_t)TT * HNH * 4);
  float*          scano = (float*)         take((size_t)TT * HNH * 4);
  unsigned short* yb    = (unsigned short*)take((size_t)TT * HNH * 2);
  float*          tmp   = (float*)         take((size_t)TT * 128 * 4);

  // 1) x = rmsnorm(x_in, ln1); bf16 copy; tap x[:, -1]
  rmsnorm_rows<<<TT, 256, 0, stream>>>(x_in, ln1, x, xb, o_xlast);

  // 2) r/k/v projections (WMMA, weight-BW bound)
  gemm_bf16_wmma<<<HNH / 64, 256, 0, stream>>>(xb, R_, Rb, nullptr, rbuf, HNH);
  gemm_bf16_wmma<<<KVD / 64, 256, 0, stream>>>(xb, K_, Kb, nullptr, kraw, KVD);
  gemm_bf16_wmma<<<KVD / 64, 256, 0, stream>>>(xb, V_, Vb, nullptr, vraw, KVD);

  // 3) per-head rmsnorm + rope
  headnorm_rope<<<TT * HH,   128, 0, stream>>>(rbuf, ln_r, cpos, HH);
  headnorm_rope<<<TT * KVHN, 128, 0, stream>>>(kraw, ln_k, cpos, KVHN);

  // 4) LoRA branches
  lora1<<<TT, 64, 0, stream>>>(x, w1, tmp, 64, 1);                       // tanh
  lora2<<<dim3(TT, HNH / 256), 256, 0, stream>>>(tmp, w2, w0, wbuf, 64, 0);
  lora1<<<TT, 64, 0, stream>>>(x, a1, tmp, 64, 0);
  lora2<<<dim3(TT, HNH / 256), 256, 0, stream>>>(tmp, a2, a0, abuf, 64, 1);
  lora1<<<TT, 128, 0, stream>>>(x, g1, tmp, 128, 2);                     // sigmoid
  lora2<<<dim3(TT, HNH / 256), 256, 0, stream>>>(tmp, g2, nullptr, gbuf, 128, 2);
  lora1<<<TT, 32, 0, stream>>>(x, v1, tmp, 32, 0);
  lora2<<<dim3(TT, HNH / 256), 256, 0, stream>>>(tmp, v2, v0, gatbf, 32, 1);

  // 5) scan prep
  prep_scan<<<TT * HH, 128, 0, stream>>>(kraw, vraw, vfirst, gatbf, wbuf, abuf,
                                         kscan, vbuf, kkbuf, dcay);

  // 6) sequential scan over T, per-head state in registers
  scan_kernel<<<HH, 128, 0, stream>>>(rbuf, dcay, kscan, vbuf, kkbuf, abuf,
                                      state0, scano, o_sfin);

  // 7) bonus + gate, pack bf16
  bonus_gate<<<TT * HH, 128, 0, stream>>>(scano, rbuf, kscan, vbuf, r_k, gbuf, yb);

  // 8) x_res = x_in + (xx*g) @ O_^T   (WMMA, residual fused)
  gemm_bf16_wmma<<<HNH / 64, 256, 0, stream>>>(yb, O_, nullptr, x_in, o_xres, HNH);

  // 9) out = rmsnorm(x_res, ln2)
  rmsnorm_rows<<<TT, 256, 0, stream>>>(o_xres, ln2, o_out, nullptr, nullptr);

  // 10) v_first passthrough (LAYER_ID != 0)
  hipMemcpyAsync(o_vfst, vfirst, (size_t)TT * HNH * 4, hipMemcpyDeviceToDevice, stream);
}
